// BaselineAttention_23338852286722
// MI455X (gfx1250) — matrix operand order repaired and checked by reading, not compiled, not run on hardware
//
#include <hip/hip_runtime.h>
#include <hip/hip_bf16.h>

typedef __attribute__((ext_vector_type(16))) _Float16 v16h;
typedef __attribute__((ext_vector_type(8)))  _Float16 v8h;
typedef __attribute__((ext_vector_type(8)))  float    v8f;

#define B_   4
#define S_   2048
#define D_   1024
#define H_   8
#define DQK_ 64
#define DH_  128
#define NTILE_KV (S_ / 16)          // 128 kv tiles per row-block

__device__ __forceinline__ v8f wmma16(v16h a, v16h b, v8f c) {
  // D = A(16x32 f16) * B(32x16 f16) + C(16x16 f32)
  return __builtin_amdgcn_wmma_f32_16x16x32_f16(false, a, false, b, (short)0, c,
                                                false, false);
}

// ---- fragment builders --------------------------------------------------
// A-fragment per ISA layout: lane&15 = M-row; e in 0..7 -> k = p[e],
// e in 8..15 -> k = p[16 + (e-8)] (caller bakes kk + 8*hi into p).
__device__ __forceinline__ v16h cvt_afrag_f32(const float* __restrict__ p) {
  v16h r;
#pragma unroll
  for (int e = 0; e < 8; ++e) {
    r[e]     = (_Float16)p[e];
    r[8 + e] = (_Float16)p[16 + e];
  }
  return r;
}
__device__ __forceinline__ v16h load_afrag_h(const _Float16* __restrict__ p) {
  v8h lo = *(const v8h*)p;
  v8h up = *(const v8h*)(p + 16);
  v16h r;
#pragma unroll
  for (int e = 0; e < 8; ++e) { r[e] = lo[e]; r[8 + e] = up[e]; }
  return r;
}
// B-fragment, ISA K order: e in 0..7 -> p[e], e in 8..15 -> p[16+(e-8)] (caller bakes 8*hi into p).
__device__ __forceinline__ v16h load_bfrag_h(const _Float16* __restrict__ p) {
  return load_afrag_h(p);               // two global_load_b128
}
__device__ __forceinline__ v16h cvt_bfrag_strided_f32(const float* __restrict__ p,
                                                      int stride) {
  v16h r;
#pragma unroll
  for (int e = 0; e < 16; ++e) r[e] = (_Float16)p[(size_t)(e < 8 ? e : e + 8) * stride];
  return r;
}

// Reductions across the 16-lane half-wave (C/D fragment column group).
__device__ __forceinline__ float redmax16(float x) {
#pragma unroll
  for (int m = 1; m < 16; m <<= 1) x = fmaxf(x, __shfl_xor(x, m, 32));
  return x;
}
__device__ __forceinline__ float redsum16(float x) {
#pragma unroll
  for (int m = 1; m < 16; m <<= 1) x += __shfl_xor(x, m, 32);
  return x;
}

// ---------------------------------------------------------------------------
// C[M,N] = A[M,K] @ W[K,N] + bias[N]  (f16 WMMA, f32 accumulate)
// TIn: float (external activations) or _Float16 (f16 scratch).
// OMODE: 0 = f32 row-major, 1 = f16 row-major, 2 = f16 per-head transposed
//        (V projection: store to [B,H,DH,S] so attn.V B-fragments are
//         contiguous b128 loads).
// block = 256 threads = 8 waves -> 32x64 tile; one 16x16 tile per wave.
// ---------------------------------------------------------------------------
template <typename TIn, int OMODE>
__global__ __launch_bounds__(256) void gemm_bias_wmma(
    const TIn* __restrict__ A, const float* __restrict__ W,
    const float* __restrict__ bias, void* __restrict__ Cout,
    int M, int N, int K) {
  const int lane = threadIdx.x & 31;
  const int wave = threadIdx.x >> 5;
  const int hi   = lane >> 4;
  const int col  = lane & 15;
  const int m0 = blockIdx.x * 32 + (wave >> 2) * 16;
  const int n0 = blockIdx.y * 64 + (wave & 3) * 16;
  const int arow = m0 + col;   // A-fragment row for this lane
  const int ncol = n0 + col;   // B/C column for this lane

  v8f acc;
  const float bv = bias[ncol];
#pragma unroll
  for (int r = 0; r < 8; ++r) acc[r] = bv;

  for (int kk = 0; kk < K; kk += 32) {
    v16h af;
    if constexpr (sizeof(TIn) == 4)
      af = cvt_afrag_f32((const float*)A + (size_t)arow * K + kk + hi * 8);
    else
      af = load_afrag_h((const _Float16*)A + (size_t)arow * K + kk + hi * 8);
    v16h bf = cvt_bfrag_strided_f32(W + (size_t)(kk + hi * 8) * N + ncol, N);
    if (kk + 32 < K) {  // prefetch next K-step (global_prefetch_b8)
      __builtin_prefetch(W + (size_t)(kk + 32 + hi * 16) * N + ncol, 0, 1);
      __builtin_prefetch(A + (size_t)arow * K + kk + 32 + hi * 8, 0, 1);
    }
    acc = wmma16(af, bf, acc);
  }

  if constexpr (OMODE == 0) {
    float* cp = (float*)Cout + (size_t)(m0 + hi * 8) * N + ncol;
#pragma unroll
    for (int r = 0; r < 8; ++r) cp[(size_t)r * N] = acc[r];
  } else if constexpr (OMODE == 1) {
    _Float16* cp = (_Float16*)Cout + (size_t)(m0 + hi * 8) * N + ncol;
#pragma unroll
    for (int r = 0; r < 8; ++r) cp[(size_t)r * N] = (_Float16)acc[r];
  } else {  // OMODE == 2: V^T per head [B,H,DH,S]
    const int h = ncol / DH_, d = ncol % DH_;
    _Float16* cp = (_Float16*)Cout;
#pragma unroll
    for (int r = 0; r < 8; ++r) {
      const int m = m0 + hi * 8 + r;
      const int b = m / S_, s = m % S_;
      cp[((size_t)(b * H_ + h) * DH_ + d) * S_ + s] = (_Float16)acc[r];
    }
  }
}

// ---------------------------------------------------------------------------
// Pass 1: per (b,h,q-tile) wave: causal online softmax stats (row max / sum).
// Q,K: f16 scratch, layout [(b*S+s)*512 + h*64 + d].
// ---------------------------------------------------------------------------
__global__ __launch_bounds__(256) void attn_pass1(
    const _Float16* __restrict__ Q, const _Float16* __restrict__ Kp,
    float* __restrict__ Ms, float* __restrict__ Ss) {
  const int lane = threadIdx.x & 31;
  const int wave = threadIdx.x >> 5;
  const int task = blockIdx.x * 8 + wave;       // B_*H_*(S_/16) tasks
  const int qt = task & (NTILE_KV - 1);
  const int h  = (task >> 7) & (H_ - 1);
  const int b  = task >> 10;
  const int hi = lane >> 4, col = lane & 15;
  const int q0 = qt * 16;
  const float scale = 0.08838834764831845f;     // 1/sqrt(128)

  const _Float16* qp = Q + ((size_t)(b * S_ + q0 + col) * (D_ / 2)) + h * DQK_;
  v16h qa0 = load_afrag_h(qp + hi * 8);
  v16h qa1 = load_afrag_h(qp + 32 + hi * 8);

  float m[8], s[8];
#pragma unroll
  for (int r = 0; r < 8; ++r) { m[r] = -3.0e38f; s[r] = 0.f; }

  for (int t = 0; t <= qt; ++t) {
    const int kv0 = t * 16;
    const _Float16* kp =
        Kp + ((size_t)(b * S_ + kv0 + col) * (D_ / 2)) + h * DQK_;
    v16h kb0 = load_bfrag_h(kp + hi * 8);
    v16h kb1 = load_bfrag_h(kp + 32 + hi * 8);
    v8f c = {};
    c = wmma16(qa0, kb0, c);
    c = wmma16(qa1, kb1, c);
#pragma unroll
    for (int r = 0; r < 8; ++r) {
      float l = c[r] * scale;
      if (kv0 + col > q0 + r + 8 * hi) l -= 1e9f;   // causal additive mask
      float nm = fmaxf(m[r], redmax16(l));
      float e  = __expf(l - nm);
      s[r] = s[r] * __expf(m[r] - nm) + redsum16(e);
      m[r] = nm;
    }
  }
  if (col == 0) {
    const size_t base = (size_t)(b * H_ + h) * S_ + q0 + 8 * hi;
#pragma unroll
    for (int r = 0; r < 8; ++r) { Ms[base + r] = m[r]; Ss[base + r] = s[r]; }
  }
}

// ---------------------------------------------------------------------------
// Pass 2: recompute logits, write normalized attn_w once (incl. zeros past
// the diagonal), fuse attn @ V via LDS f16 staging; causal-skip AV pairs.
// V is f16 [B,H,DH,S] (per-head transposed) -> contiguous B-fragments.
// Output attn-out as f16 [B,S,D] for the fc GEMM.
// ---------------------------------------------------------------------------
__global__ __launch_bounds__(256) void attn_pass2(
    const _Float16* __restrict__ Q, const _Float16* __restrict__ Kp,
    const _Float16* __restrict__ Vt, const float* __restrict__ Ms,
    const float* __restrict__ Ss, float* __restrict__ AttnW,
    _Float16* __restrict__ AttnOut) {
  __shared__ __align__(16) _Float16 lds[8 * 16 * 32];  // 16x32 f16 per wave
  const int lane = threadIdx.x & 31;
  const int wave = threadIdx.x >> 5;
  const int task = blockIdx.x * 8 + wave;
  const int qt = task & (NTILE_KV - 1);
  const int h  = (task >> 7) & (H_ - 1);
  const int b  = task >> 10;
  const int hi = lane >> 4, col = lane & 15;
  const int q0 = qt * 16;
  const float scale = 0.08838834764831845f;
  _Float16* lp = &lds[wave * 16 * 32];

  const _Float16* qp = Q + ((size_t)(b * S_ + q0 + col) * (D_ / 2)) + h * DQK_;
  v16h qa0 = load_afrag_h(qp + hi * 8);
  v16h qa1 = load_afrag_h(qp + 32 + hi * 8);

  float m[8], is[8];
  {
    const size_t base = (size_t)(b * H_ + h) * S_ + q0 + 8 * hi;
#pragma unroll
    for (int r = 0; r < 8; ++r) { m[r] = Ms[base + r]; is[r] = 1.f / Ss[base + r]; }
  }

  v8f acc[8];
#pragma unroll
  for (int dt = 0; dt < 8; ++dt) acc[dt] = (v8f){};

  const size_t awrow = (size_t)(b * H_ + h) * S_ + q0 + 8 * hi;  // + r rows
  const size_t vhead = (size_t)(b * H_ + h) * DH_;               // + d rows

  for (int t = 0; t <= qt; ++t) {
    const int kv0 = t * 16;
    const _Float16* kp =
        Kp + ((size_t)(b * S_ + kv0 + col) * (D_ / 2)) + h * DQK_;
    v16h kb0 = load_bfrag_h(kp + hi * 8);
    v16h kb1 = load_bfrag_h(kp + 32 + hi * 8);
    v8f c = {};
    c = wmma16(qa0, kb0, c);
    c = wmma16(qa1, kb1, c);

    float p[8];
#pragma unroll
    for (int r = 0; r < 8; ++r) {
      float l = c[r] * scale;
      if (kv0 + col > q0 + r + 8 * hi) l -= 1e9f;
      p[r] = __expf(l - m[r]) * is[r];
    }

    // softmax output (each element written exactly once)
#pragma unroll
    for (int r = 0; r < 8; ++r)
      AttnW[(awrow + r) * S_ + kv0 + col] = p[r];

    // stage probability tile into LDS as f16 (wave-local, DS in-order)
    const int half = (t & 1) * 16;
#pragma unroll
    for (int r = 0; r < 8; ++r)
      lp[(r + 8 * hi) * 32 + half + col] = (_Float16)p[r];

    const bool last = (t == qt);
    if ((t & 1) || last) {
      if (last && !(t & 1)) {  // zero-pad the unused half of the last pair
#pragma unroll
        for (int r = 0; r < 8; ++r)
          lp[(r + 8 * hi) * 32 + 16 + col] = (_Float16)0.f;
      }
      // A-fragment (16 q-rows x K=32 kv) from LDS: two ds_load_b128
      v8h pa0 = *(const v8h*)&lp[col * 32 + hi * 8];
      v8h pa1 = *(const v8h*)&lp[col * 32 + 16 + hi * 8];
      v16h pa;
#pragma unroll
      for (int e = 0; e < 8; ++e) { pa[e] = pa0[e]; pa[8 + e] = pa1[e]; }
      const int kvb = (t & ~1) * 16;
#pragma unroll
      for (int dt = 0; dt < 8; ++dt) {
        // B-fragment from V^T: lane col = d, elements = contiguous kv
        v16h vb = load_bfrag_h(
            Vt + (vhead + dt * 16 + col) * S_ + kvb + hi * 8);
        acc[dt] = wmma16(pa, vb, acc[dt]);
      }
    }
  }

  // masked region of attn_w is exactly zero
  for (int t = qt + 1; t < NTILE_KV; ++t) {
#pragma unroll
    for (int r = 0; r < 8; ++r)
      AttnW[(awrow + r) * S_ + t * 16 + col] = 0.f;
  }

  // attn out (f16) in [B,S,D] layout (b, s, h*128 + d) for the fc GEMM
#pragma unroll
  for (int dt = 0; dt < 8; ++dt) {
#pragma unroll
    for (int r = 0; r < 8; ++r) {
      AttnOut[(size_t)(b * S_ + q0 + r + 8 * hi) * D_ + h * DH_ + dt * 16 + col] =
          (_Float16)acc[dt][r];
    }
  }
}

extern "C" void kernel_launch(void* const* d_in, const int* in_sizes, int n_in,
                              void* d_out, int out_size, void* d_ws, size_t ws_size,
                              hipStream_t stream) {
  (void)in_sizes; (void)n_in; (void)out_size; (void)ws_size;
  const float* q_in = (const float*)d_in[0];
  const float* k_in = (const float*)d_in[1];
  const float* v_in = (const float*)d_in[2];
  /* d_in[3] = boolean causal mask: applied analytically, not read */
  const float* Wq_w = (const float*)d_in[4];
  const float* Wq_b = (const float*)d_in[5];
  const float* Wk_w = (const float*)d_in[6];
  const float* Wk_b = (const float*)d_in[7];
  const float* Wv_w = (const float*)d_in[8];
  const float* Wv_b = (const float*)d_in[9];
  const float* fc_w = (const float*)d_in[10];
  const float* fc_b = (const float*)d_in[11];

  float* out   = (float*)d_out;                       // [B,S,D] f32
  float* attnw = out + (size_t)B_ * S_ * D_;          // [B,H,S,S] f32

  _Float16* qw = (_Float16*)d_ws;                     // [B*S, 512]   f16
  _Float16* kw = qw + (size_t)B_ * S_ * (D_ / 2);     // [B*S, 512]   f16
  _Float16* vt = kw + (size_t)B_ * S_ * (D_ / 2);     // [B,H,128,S]  f16
  _Float16* ao = vt + (size_t)B_ * S_ * D_;           // [B*S, 1024]  f16
  float* msb = (float*)(ao + (size_t)B_ * S_ * D_);   // [B,H,S]      f32
  float* ssb = msb + (size_t)B_ * H_ * S_;            // [B,H,S]      f32

  const int M = B_ * S_;          // 8192
  dim3 blk(256);

  gemm_bias_wmma<float, 1><<<dim3(M / 32, (D_ / 2) / 64), blk, 0, stream>>>(
      q_in, Wq_w, Wq_b, qw, M, D_ / 2, D_);
  gemm_bias_wmma<float, 1><<<dim3(M / 32, (D_ / 2) / 64), blk, 0, stream>>>(
      k_in, Wk_w, Wk_b, kw, M, D_ / 2, D_);
  gemm_bias_wmma<float, 2><<<dim3(M / 32, D_ / 64), blk, 0, stream>>>(
      v_in, Wv_w, Wv_b, vt, M, D_, D_);

  const int ntask = B_ * H_ * (S_ / 16);   // 4096 wave-tasks, 8 waves/block
  attn_pass1<<<dim3(ntask / 8), blk, 0, stream>>>(qw, kw, msb, ssb);
  attn_pass2<<<dim3(ntask / 8), blk, 0, stream>>>(qw, kw, vt, msb, ssb, attnw, ao);

  gemm_bias_wmma<_Float16, 0><<<dim3(M / 32, D_ / 64), blk, 0, stream>>>(
      ao, fc_w, fc_b, out, M, D_, D_);
}
